// DiscreteGNNPolicy_61624190763734
// MI455X (gfx1250) — matrix operand order repaired and checked by reading, not compiled, not run on hardware
//
#include <hip/hip_runtime.h>
#include <hip/hip_bf16.h>

#define NEG_SLOPE 0.2f

typedef __attribute__((ext_vector_type(16))) __bf16 v16bf;
typedef __attribute__((ext_vector_type(8)))  float  v8f;

union Frag { v16bf v; uint4 q[2]; unsigned u[8]; };

__device__ __forceinline__ unsigned short f2bf(float f) {
  unsigned u = __float_as_uint(f);
  u += 0x7fffu + ((u >> 16) & 1u);           // round to nearest even
  return (unsigned short)(u >> 16);
}
__device__ __forceinline__ unsigned pack2bf(float a, float b) {
  return (unsigned)f2bf(a) | ((unsigned)f2bf(b) << 16);
}
// order-preserving float<->uint for atomicMax on signed floats
__device__ __forceinline__ unsigned fenc(float f) {
  unsigned u = __float_as_uint(f);
  return (u >> 31) ? ~u : (u | 0x80000000u);
}
__device__ __forceinline__ float fdec(unsigned u) {
  return __uint_as_float((u >> 31) ? (u & 0x7fffffffu) : ~u);
}

// 4-word-granular xor swizzle: touches only index bits [5:2], so aligned runs
// of 4 words stay contiguous (-> ds_load_b128) while different rows map their
// 4-word groups onto different bank quartets (16 rows x 4 banks = all 64 banks).
__device__ __forceinline__ int swz4(int row, int kp) { return kp ^ ((row << 2) & 60); }

__global__ void k_zero(unsigned* __restrict__ p, int n) {
  for (int i = blockIdx.x * blockDim.x + threadIdx.x; i < n; i += gridDim.x * blockDim.x)
    p[i] = 0u;
}

// ---- fold lew (L,128,128) with a_e (L,4,32) -> M (L,128,4) ----
__global__ void k_prepM(const float* __restrict__ lew, const float* __restrict__ ae,
                        float* __restrict__ M) {
  int idx = blockIdx.x * blockDim.x + threadIdx.x;
  if (idx >= 3 * 128 * 4) return;
  int l = idx >> 9, r = idx & 511, k = r >> 2, h = r & 3;
  float s = 0.f;
  const float* wrow = lew + l * 16384 + k * 128 + h * 32;
  const float* arow = ae + l * 128 + h * 32;
#pragma unroll 8
  for (int c = 0; c < 32; ++c) s += wrow[c] * arow[c];
  M[idx] = s;   // layout [l][k][h]
}

// ---- fold ep_w/ep_b through M -> em (L,2,4), ebl (L,4) ----
__global__ void k_prep2(const float* __restrict__ epw, const float* __restrict__ epb,
                        const float* __restrict__ M, float* __restrict__ em,
                        float* __restrict__ ebl) {
  int t = threadIdx.x;
  if (t < 24) {
    int l = t / 8, i = (t >> 2) & 1, h = t & 3;
    float s = 0.f;
    for (int k = 0; k < 128; ++k) s += epw[i * 128 + k] * M[l * 512 + k * 4 + h];
    em[(l * 2 + i) * 4 + h] = s;
  } else if (t >= 32 && t < 44) {
    int t2 = t - 32, l = t2 >> 2, h = t2 & 3;
    float s = 0.f;
    for (int k = 0; k < 128; ++k) s += epb[k] * M[l * 512 + k * 4 + h];
    ebl[l * 4 + h] = s;
  }
}

// ---- x = node_features @ np_w + np_b ----
__global__ void k_node_proj(const float* __restrict__ nf, const float* __restrict__ w,
                            const float* __restrict__ b, float* __restrict__ x, int N) {
  int idx = blockIdx.x * blockDim.x + threadIdx.x;
  if (idx >= N * 128) return;
  int n = idx >> 7, j = idx & 127;
  x[idx] = nf[n * 3] * w[j] + nf[n * 3 + 1] * w[128 + j] + nf[n * 3 + 2] * w[256 + j] + b[j];
}

// ---- per-dst sums of edge attention contributions (all 3 layers) + counts ----
__global__ void k_loop_contrib(const int* __restrict__ ei, const float* __restrict__ ea,
                               const float* __restrict__ em, const float* __restrict__ ebl,
                               float* __restrict__ loopc, float* __restrict__ cnt, int E) {
  int e = blockIdx.x * blockDim.x + threadIdx.x;
  if (e >= E) return;
  int d = ei[E + e];
  float a0 = ea[2 * e], a1 = ea[2 * e + 1];
#pragma unroll
  for (int l = 0; l < 3; ++l)
#pragma unroll
    for (int h = 0; h < 4; ++h) {
      float c = a0 * em[(l * 2) * 4 + h] + a1 * em[(l * 2 + 1) * 4 + h] + ebl[l * 4 + h];
      atomicAdd(&loopc[d * 12 + l * 4 + h], c);
    }
  atomicAdd(&cnt[d], 1.f);
}

__global__ void k_loop_norm(float* __restrict__ loopc, const float* __restrict__ cnt, int N) {
  int idx = blockIdx.x * blockDim.x + threadIdx.x;
  if (idx >= N * 12) return;
  loopc[idx] /= fmaxf(cnt[idx / 12], 1.f);
}

// ---- WMMA bf16 GEMM: Out[nrows,128] = act(A[nrows,128] @ W[128,128] + bias) ----
// 8 waves/block; wave w owns rows [bid*128 + 16w, +16), all 128 cols.
__global__ __launch_bounds__(256) void k_gemm128(
    const float* __restrict__ A, const float* __restrict__ W,
    const float* __restrict__ bias, float* __restrict__ Out, int nrows, int act) {
  __shared__ __attribute__((aligned(16))) unsigned As[128 * 64];  // A tile, bf16 pairs
  __shared__ __attribute__((aligned(16))) unsigned Ws[128 * 64];  // W^T: [j][kp]
  const int tid = threadIdx.x;
  const int lane = tid & 31, wave = tid >> 5;
  const int r0 = blockIdx.x * 128;
  // W^T load (coalesced over j)
  for (int p = tid; p < 128 * 64; p += 256) {
    int kp = p >> 7, j = p & 127;
    unsigned pk = pack2bf(W[(2 * kp) * 128 + j], W[(2 * kp + 1) * 128 + j]);
    Ws[j * 64 + swz4(j, kp)] = pk;
  }
  // A tile load (coalesced over k)
  for (int p = tid; p < 128 * 64; p += 256) {
    int row = p >> 6, kp = p & 63;
    int gr = r0 + row; if (gr >= nrows) gr = 0;   // clamp; results masked at store
    const float* s = A + (size_t)gr * 128 + 2 * kp;
    As[row * 64 + swz4(row, kp)] = pack2bf(s[0], s[1]);
  }
  __syncthreads();

  const int rt = wave;
  const int hi = lane >> 4;         // 0: lanes 0-15, 1: lanes 16-31
  const int l15 = lane & 15;
  v8f acc[8];
#pragma unroll
  for (int i = 0; i < 8; ++i) acc[i] = (v8f){0.f,0.f,0.f,0.f,0.f,0.f,0.f,0.f};

  const int arow = rt * 16 + l15;
  const unsigned* Arow = &As[arow * 64];
  const int aswz = (arow << 2) & 60;
#pragma unroll
  for (int ks = 0; ks < 4; ++ks) {       // K = 128 in 4 steps of 32
    const int kpb = ks * 16;
    // 16-bit A layout: lanes<16 hold K0-7 & K16-23; lanes>=16 hold K8-15 & K24-31
    // -> two aligned 4-word runs per fragment, loaded as ds_load_b128
    Frag a;
    a.q[0] = *(const uint4*)&Arow[(kpb + (hi ? 4 : 0)) ^ aswz];
    a.q[1] = *(const uint4*)&Arow[(kpb + (hi ? 12 : 8)) ^ aswz];
#pragma unroll
    for (int ct = 0; ct < 8; ++ct) {
      const int bcol = ct * 16 + l15;
      const unsigned* Brow = &Ws[bcol * 64];
      const int bswz = (bcol << 2) & 60;
      const int b0 = kpb + (hi ? 4 : 0);  // B layout: same as A (K0-7/16-23 lo, K8-15/24-31 hi)
      Frag b;
      b.q[0] = *(const uint4*)&Brow[b0 ^ bswz];
      b.q[1] = *(const uint4*)&Brow[(b0 + 8) ^ bswz];
      acc[ct] = __builtin_amdgcn_wmma_f32_16x16x32_bf16(
          false, a.v, false, b.v, (short)0, acc[ct], false, false);
    }
  }
#pragma unroll
  for (int ct = 0; ct < 8; ++ct) {
    int col = ct * 16 + l15;
    float bv = bias ? bias[col] : 0.f;
#pragma unroll
    for (int r = 0; r < 8; ++r) {
      int row = r0 + rt * 16 + r + (hi ? 8 : 0);   // D layout: M = r + 8*hi
      if (row < nrows) {
        float v = acc[ct][r] + bv;
        if (act == 1) v = fmaxf(v, 0.f);
        Out[(size_t)row * 128 + col] = v;
      }
    }
  }
}

// ---- a_src/a_dst: per node, per head dot of xs with attention vectors ----
__global__ void k_att_nodes(const float* __restrict__ xs, const float* __restrict__ aw_s,
                            const float* __restrict__ aw_d, float* __restrict__ a_src,
                            float* __restrict__ a_dst, int N) {
  int j = threadIdx.x;          // 0..127, head = j>>5 (wave32 == one head)
  int h = j >> 5, lane = j & 31;
  float ws = aw_s[j], wd = aw_d[j];
  for (int n = blockIdx.x; n < N; n += gridDim.x) {
    float v = xs[(size_t)n * 128 + j];
    float s = v * ws, d = v * wd;
#pragma unroll
    for (int o = 16; o; o >>= 1) { s += __shfl_down(s, o, 32); d += __shfl_down(d, o, 32); }
    if (lane == 0) { a_src[n * 4 + h] = s; a_dst[n * 4 + h] = d; }
  }
}

__device__ __forceinline__ float edge_raw(int e, int h, int l, int E,
                                          const int* ei, const float* ea,
                                          const float* em, const float* ebl,
                                          const float* loopc, const float* a_src,
                                          const float* a_dst, int* s_out, int* d_out_) {
  int s, d; float c;
  if (e < E) {
    s = ei[e]; d = ei[E + e];
    c = ea[2 * e] * em[(l * 2) * 4 + h] + ea[2 * e + 1] * em[(l * 2 + 1) * 4 + h] + ebl[l * 4 + h];
  } else {
    int n = e - E; s = d = n; c = loopc[n * 12 + l * 4 + h];
  }
  *s_out = s; *d_out_ = d;
  float raw = a_src[s * 4 + h] + a_dst[d * 4 + h] + c;
  return raw > 0.f ? raw : NEG_SLOPE * raw;
}

__global__ void k_att_max(const int* __restrict__ ei, const float* __restrict__ ea,
                          const float* __restrict__ em, const float* __restrict__ ebl,
                          const float* __restrict__ loopc, const float* __restrict__ a_src,
                          const float* __restrict__ a_dst, unsigned* __restrict__ maxenc,
                          int l, int E, int E2) {
  int e = blockIdx.x * blockDim.x + threadIdx.x;
  if (e >= E2) return;
#pragma unroll
  for (int h = 0; h < 4; ++h) {
    int s, d;
    float raw = edge_raw(e, h, l, E, ei, ea, em, ebl, loopc, a_src, a_dst, &s, &d);
    atomicMax(&maxenc[d * 4 + h], fenc(raw));
  }
}

__global__ void k_att_sum(const int* __restrict__ ei, const float* __restrict__ ea,
                          const float* __restrict__ em, const float* __restrict__ ebl,
                          const float* __restrict__ loopc, const float* __restrict__ a_src,
                          const float* __restrict__ a_dst, const unsigned* __restrict__ maxenc,
                          float* __restrict__ sums, int l, int E, int E2) {
  int e = blockIdx.x * blockDim.x + threadIdx.x;
  if (e >= E2) return;
#pragma unroll
  for (int h = 0; h < 4; ++h) {
    int s, d;
    float raw = edge_raw(e, h, l, E, ei, ea, em, ebl, loopc, a_src, a_dst, &s, &d);
    atomicAdd(&sums[d * 4 + h], __expf(raw - fdec(maxenc[d * 4 + h])));
  }
}

// ---- aggregation: one wave per edge, float4 per lane, atomics to acc[dst] ----
__global__ __launch_bounds__(256) void k_att_agg(
    const int* __restrict__ ei, const float* __restrict__ ea,
    const float* __restrict__ em, const float* __restrict__ ebl,
    const float* __restrict__ loopc, const float* __restrict__ a_src,
    const float* __restrict__ a_dst, const unsigned* __restrict__ maxenc,
    const float* __restrict__ sums, const float* __restrict__ xs,
    float* __restrict__ acc, int l, int E, int E2) {
  int e = blockIdx.x * 8 + (threadIdx.x >> 5);
  if (e >= E2) return;
  int lane = threadIdx.x & 31;
  int h = lane >> 3;                          // 8 lanes (32 channels) per head
  int s, d;
  float raw = edge_raw(e, h, l, E, ei, ea, em, ebl, loopc, a_src, a_dst, &s, &d);
  float p = __expf(raw - fdec(maxenc[d * 4 + h])) / sums[d * 4 + h];
  const float4 xv = *(const float4*)(xs + (size_t)s * 128 + lane * 4);
  float* ar = acc + (size_t)d * 128 + lane * 4;
  atomicAdd(ar + 0, xv.x * p);
  atomicAdd(ar + 1, xv.y * p);
  atomicAdd(ar + 2, xv.z * p);
  atomicAdd(ar + 3, xv.w * p);
}

__global__ void k_bias_elu(const float* __restrict__ acc, const float* __restrict__ bias,
                           float* __restrict__ x, float* __restrict__ emb, int N) {
  int idx = blockIdx.x * blockDim.x + threadIdx.x;
  if (idx >= N * 128) return;
  float v = acc[idx] + bias[idx & 127];
  float r = v > 0.f ? v : expm1f(v);
  x[idx] = r;
  if (emb) emb[idx] = r;
}

__global__ void k_action(const float* __restrict__ h1, const float* __restrict__ w2,
                         const float* __restrict__ b2, float* __restrict__ out, int N) {
  int idx = blockIdx.x * blockDim.x + threadIdx.x;
  if (idx >= N * 8) return;
  int n = idx >> 3, o = idx & 7;
  float a = b2[o];
  const float* hr = h1 + (size_t)n * 128;
#pragma unroll 8
  for (int j = 0; j < 128; ++j) a += hr[j] * w2[j * 8 + o];
  out[idx] = a;
}

__global__ void k_meanpool(const float* __restrict__ x, float* __restrict__ gsum, int N) {
  int j = threadIdx.x;
  float s = 0.f;
  for (int n = blockIdx.x; n < N; n += gridDim.x) s += x[(size_t)n * 128 + j];
  atomicAdd(&gsum[j], s);
}

__global__ void k_value(const float* __restrict__ gsum, const float* __restrict__ w1,
                        const float* __restrict__ b1, const float* __restrict__ w2,
                        const float* __restrict__ b2, float* __restrict__ out, float invN) {
  __shared__ float ge[128];
  __shared__ float rr[128];
  int j = threadIdx.x;
  ge[j] = gsum[j] * invN;
  __syncthreads();
  float a = b1[j];
  for (int k = 0; k < 128; ++k) a += ge[k] * w1[k * 128 + j];
  float g = fmaxf(a, 0.f);
  rr[j] = g * w2[j];
  __syncthreads();
  for (int s = 64; s; s >>= 1) { if (j < s) rr[j] += rr[j + s]; __syncthreads(); }
  if (j == 0) out[0] = rr[0] + b2[0];
}

extern "C" void kernel_launch(void* const* d_in, const int* in_sizes, int n_in,
                              void* d_out, int out_size, void* d_ws, size_t ws_size,
                              hipStream_t stream) {
  const int N = in_sizes[0] / 3;
  const int E = in_sizes[2] / 2;
  const int E2 = E + N;

  const float* nf    = (const float*)d_in[0];
  const int*   ei    = (const int*)  d_in[1];
  const float* eattr = (const float*)d_in[2];
  const float* np_w  = (const float*)d_in[3];
  const float* np_b  = (const float*)d_in[4];
  const float* ep_w  = (const float*)d_in[5];
  const float* ep_b  = (const float*)d_in[6];
  const float* lw    = (const float*)d_in[7];
  const float* lew   = (const float*)d_in[8];
  const float* aw_s  = (const float*)d_in[9];
  const float* aw_d  = (const float*)d_in[10];
  const float* aw_e  = (const float*)d_in[11];
  const float* gbias = (const float*)d_in[12];
  const float* ah_w1 = (const float*)d_in[13];
  const float* ah_b1 = (const float*)d_in[14];
  const float* ah_w2 = (const float*)d_in[15];
  const float* ah_b2 = (const float*)d_in[16];
  const float* vh_w1 = (const float*)d_in[17];
  const float* vh_b1 = (const float*)d_in[18];
  const float* vh_w2 = (const float*)d_in[19];
  const float* vh_b2 = (const float*)d_in[20];

  float* out        = (float*)d_out;
  float* out_logits = out;
  float* out_value  = out + (size_t)N * 8;
  float* out_emb    = out + (size_t)N * 8 + 1;

  // workspace carve-up
  char* w = (char*)d_ws;
  size_t off = 0;
  auto carve = [&](size_t bytes) { void* p = w + off; off += (bytes + 255) & ~(size_t)255; return p; };
  float*    x      = (float*)   carve((size_t)N * 128 * 4);
  float*    xs     = (float*)   carve((size_t)N * 128 * 4);
  float*    acc    = (float*)   carve((size_t)N * 128 * 4);
  float*    a_src  = (float*)   carve((size_t)N * 4 * 4);
  float*    a_dst  = (float*)   carve((size_t)N * 4 * 4);
  unsigned* maxenc = (unsigned*)carve((size_t)N * 4 * 4);
  float*    sums   = (float*)   carve((size_t)N * 4 * 4);
  float*    loopc  = (float*)   carve((size_t)N * 12 * 4);
  float*    cnt    = (float*)   carve((size_t)N * 4);
  float*    gsum   = (float*)   carve(128 * 4);
  float*    Mbuf   = (float*)   carve(1536 * 4);
  float*    em     = (float*)   carve(24 * 4);
  float*    ebl    = (float*)   carve(12 * 4);

  const int T = 256;
  auto nb = [](long n, int t) { return (unsigned)((n + t - 1) / t); };

  // fold edge-attention weights
  k_prepM<<<nb(1536, T), T, 0, stream>>>(lew, aw_e, Mbuf);
  k_prep2<<<1, 64, 0, stream>>>(ep_w, ep_b, Mbuf, em, ebl);
  // node projection
  k_node_proj<<<nb((long)N * 128, T), T, 0, stream>>>(nf, np_w, np_b, x, N);
  // self-loop contributions (mean over incoming edges)
  k_zero<<<2048, T, 0, stream>>>((unsigned*)loopc, N * 12);
  k_zero<<<2048, T, 0, stream>>>((unsigned*)cnt, N);
  k_loop_contrib<<<nb(E, T), T, 0, stream>>>(ei, eattr, em, ebl, loopc, cnt, E);
  k_loop_norm<<<nb((long)N * 12, T), T, 0, stream>>>(loopc, cnt, N);

  for (int l = 0; l < 3; ++l) {
    k_zero<<<4096, T, 0, stream>>>((unsigned*)acc, N * 128);
    k_zero<<<2048, T, 0, stream>>>(maxenc, N * 4);
    k_zero<<<2048, T, 0, stream>>>((unsigned*)sums, N * 4);
    k_gemm128<<<nb(N, 128), 256, 0, stream>>>(x, lw + (size_t)l * 16384, nullptr, xs, N, 0);
    k_att_nodes<<<2048, 128, 0, stream>>>(xs, aw_s + l * 128, aw_d + l * 128, a_src, a_dst, N);
    k_att_max<<<nb(E2, T), T, 0, stream>>>(ei, eattr, em, ebl, loopc, a_src, a_dst, maxenc, l, E, E2);
    k_att_sum<<<nb(E2, T), T, 0, stream>>>(ei, eattr, em, ebl, loopc, a_src, a_dst, maxenc, sums, l, E, E2);
    k_att_agg<<<nb(E2, 8), 256, 0, stream>>>(ei, eattr, em, ebl, loopc, a_src, a_dst, maxenc,
                                             sums, xs, acc, l, E, E2);
    k_bias_elu<<<nb((long)N * 128, T), T, 0, stream>>>(acc, gbias + l * 128, x,
                                                       (l == 2) ? out_emb : nullptr, N);
  }

  // heads
  k_gemm128<<<nb(N, 128), 256, 0, stream>>>(x, ah_w1, ah_b1, xs, N, 1);
  k_action<<<nb((long)N * 8, T), T, 0, stream>>>(xs, ah_w2, ah_b2, out_logits, N);
  k_zero<<<1, 128, 0, stream>>>((unsigned*)gsum, 128);
  k_meanpool<<<480, 128, 0, stream>>>(x, gsum, N);
  k_value<<<1, 128, 0, stream>>>(gsum, vh_w1, vh_b1, vh_w2, vh_b2, out_value, 1.0f / (float)N);
}
